// GINEncoder_71605694759276
// MI455X (gfx1250) — matrix operand order repaired and checked by reading, not compiled, not run on hardware
//
#include <hip/hip_runtime.h>
#include <hip/hip_bf16.h>
#include <stdint.h>

#define NNODES 50000
#define NEDGES 800000
#define DIN    64
#define DH     512
#define DOUTC  128

typedef __attribute__((ext_vector_type(16))) __bf16 v16bf;
typedef __attribute__((ext_vector_type(8)))  __bf16 v8bf;
typedef __attribute__((ext_vector_type(8)))  float  v8f;

__device__ __forceinline__ unsigned short f32_to_bf16(float f) {
    union { float f; unsigned u; } v; v.f = f;
    unsigned u = v.u;
    u += 0x7fffu + ((u >> 16) & 1u);      // round-to-nearest-even
    return (unsigned short)(u >> 16);
}

// ---------------- generic f32 -> bf16 convert ----------------
__global__ void k_cvt_f32_bf16(const float* __restrict__ in,
                               unsigned short* __restrict__ out, int n) {
    int i = blockIdx.x * blockDim.x + threadIdx.x;
    if (i < n) out[i] = f32_to_bf16(in[i]);
}

// ---------------- vectorized f32 copy (n multiple of 4) ----------------
__global__ void k_copy_f32x4(const float* __restrict__ in,
                             float* __restrict__ out, int n4) {
    int i = blockIdx.x * blockDim.x + threadIdx.x;
    if (i < n4) ((float4*)out)[i] = ((const float4*)in)[i];
}

// ------- edge-parallel scatter-add: dst_feat[dst] += src_feat[src] -------
// D must be a multiple of 4. One thread handles 4 channels of one edge.
__global__ void k_scatter_add4(const float* __restrict__ feat,
                               const long long* __restrict__ ei,
                               float* __restrict__ acc, int E, int D) {
    const int quads = D >> 2;
    long long t = (long long)blockIdx.x * blockDim.x + threadIdx.x;
    if (t >= (long long)E * quads) return;
    int e = (int)(t / quads);
    int q = (int)(t % quads);
    long long s = ei[e];
    long long d = ei[E + e];
    const float4 v = *(const float4*)(feat + s * (long long)D + q * 4);
    float* p = acc + d * (long long)D + q * 4;
    atomicAdd(p + 0, v.x);
    atomicAdd(p + 1, v.y);
    atomicAdd(p + 2, v.z);
    atomicAdd(p + 3, v.w);
}

// -------- fragment loader: 16-bit A/B striping per ISA 7.12.2 --------
__device__ __forceinline__ v16bf load_frag(const __bf16* p, int k, int stride2) {
    // two 16B loads: elements 0..7 at p+k, elements 8..15 at p+k+stride2
    v8bf lo = *(const v8bf*)(p + k);
    v8bf hi = *(const v8bf*)(p + k + stride2);
    return __builtin_shufflevector(lo, hi, 0,1,2,3,4,5,6,7,8,9,10,11,12,13,14,15);
}

// ---------------- WMMA GEMM: out = act(A[M,K] @ W[Dn,K]^T + bias) ----------------
// A, W bf16; f32 accumulate via v_wmma_f32_16x16x32_bf16.
// 8 waves / block arranged 4(M) x 2(N); each wave computes a 32x32 tile with
// 2x2 register blocking (4 accumulators) -> 4 WMMAs per 8 b128 fragment loads.
// act: 0 = identity, 1 = leaky-relu(0.1), 2 = relu.  Exactly one of outF/outB non-null.
__global__ __launch_bounds__(256)
void k_gemm_bias_act(const __bf16* __restrict__ A, const __bf16* __restrict__ W,
                     const float* __restrict__ bias,
                     float* __restrict__ outF, unsigned short* __restrict__ outB,
                     int M, int K, int Dn, int act)
{
    const int lane = threadIdx.x & 31;
    const int wave = threadIdx.x >> 5;
    const int m0 = blockIdx.x * 128 + (wave & 3) * 32;   // 4 waves in M
    const int n0 = blockIdx.y * 64  + (wave >> 2) * 32;  // 2 waves in N
    if (m0 >= M) return;                 // wave-uniform; M multiple of 16
    const bool m1ok = (m0 + 16) < M;     // second M-subtile fully valid?

    const int r  = lane & 15;
    // A striping: lanes 0-15 K=k+0..7 & k+16..23 ; lanes 16-31 K=k+8..15 & k+24..31
    const int kA = (lane >> 4) << 3;     // 0 or 8
    // B striping: lanes 0-15 K=k+0..7 & k+16..23 ; lanes 16-31 K=k+8..15 & k+24..31
    const int kB = (lane >> 4) << 3;     // 0 or 8

    const int r1 = m0 + 16 + r;
    const __bf16* __restrict__ pa0 = A + (size_t)(m0 + r) * K + kA;
    const __bf16* __restrict__ pa1 = A + (size_t)(r1 < M ? r1 : M - 1) * K + kA; // clamped
    const __bf16* __restrict__ pb0 = W + (size_t)(n0 + r) * K + kB;
    const __bf16* __restrict__ pb1 = W + (size_t)(n0 + 16 + r) * K + kB;

    v8f c00 = {0.f,0.f,0.f,0.f,0.f,0.f,0.f,0.f};
    v8f c01 = c00, c10 = c00, c11 = c00;

    for (int k = 0; k < K; k += 32) {
        v16bf a0 = load_frag(pa0, k, 16);
        v16bf a1 = load_frag(pa1, k, 16);
        v16bf b0 = load_frag(pb0, k, 16);
        v16bf b1 = load_frag(pb1, k, 16);
        if (k + 32 < K) {
            __builtin_prefetch(pa0 + k + 32, 0, 3);   // global_prefetch_b8
            __builtin_prefetch(pa1 + k + 32, 0, 3);
            __builtin_prefetch(pb0 + k + 32, 0, 3);
            __builtin_prefetch(pb1 + k + 32, 0, 3);
        }
        c00 = __builtin_amdgcn_wmma_f32_16x16x32_bf16(false, a0, false, b0, (short)0, c00, false, false);
        c01 = __builtin_amdgcn_wmma_f32_16x16x32_bf16(false, a0, false, b1, (short)0, c01, false, false);
        c10 = __builtin_amdgcn_wmma_f32_16x16x32_bf16(false, a1, false, b0, (short)0, c10, false, false);
        c11 = __builtin_amdgcn_wmma_f32_16x16x32_bf16(false, a1, false, b1, (short)0, c11, false, false);
    }

    // C/D layout: VGPR i -> row base+i (lanes 0-15) / base+8+i (lanes 16-31),
    // col = colbase + lane%16
    const int   rsub = (lane >> 4) << 3;     // 0 or 8
    const float bv0  = bias[n0 + r];
    const float bv1  = bias[n0 + 16 + r];

    auto store_tile = [&](const v8f& c, int rowbase, int col, float bv) {
#pragma unroll
        for (int i = 0; i < 8; ++i) {
            float v = c[i] + bv;
            if      (act == 1) v = (v > 0.f) ? v : 0.1f * v;
            else if (act == 2) v = (v > 0.f) ? v : 0.f;
            const size_t off = (size_t)(rowbase + i) * Dn + col;
            if (outF) outF[off] = v;
            else      outB[off] = f32_to_bf16(v);
        }
    };

    store_tile(c00, m0 + rsub,      n0 + r,      bv0);
    store_tile(c01, m0 + rsub,      n0 + 16 + r, bv1);
    if (m1ok) {
        store_tile(c10, m0 + 16 + rsub, n0 + r,      bv0);
        store_tile(c11, m0 + 16 + rsub, n0 + 16 + r, bv1);
    }
}

// ---------------------------------------------------------------------------

static inline size_t align256(size_t x) { return (x + 255) & ~(size_t)255; }

extern "C" void kernel_launch(void* const* d_in, const int* in_sizes, int n_in,
                              void* d_out, int out_size, void* d_ws, size_t ws_size,
                              hipStream_t stream) {
    (void)in_sizes; (void)n_in; (void)out_size; (void)ws_size;

    const float*     x    = (const float*)d_in[0];
    const long long* ei   = (const long long*)d_in[1];
    const float* W_s1 = (const float*)d_in[2];  const float* b_s1 = (const float*)d_in[3];
    const float* W_s2 = (const float*)d_in[4];  const float* b_s2 = (const float*)d_in[5];
    const float* W_s3 = (const float*)d_in[6];  const float* b_s3 = (const float*)d_in[7];
    const float* W_s4 = (const float*)d_in[8];  const float* b_s4 = (const float*)d_in[9];
    const float* W_m1 = (const float*)d_in[10]; const float* b_m1 = (const float*)d_in[11];
    const float* W_m2 = (const float*)d_in[12]; const float* b_m2 = (const float*)d_in[13];
    const float* W_l1 = (const float*)d_in[14]; const float* b_l1 = (const float*)d_in[15];
    const float* W_l2 = (const float*)d_in[16]; const float* b_l2 = (const float*)d_in[17];
    float* out = (float*)d_out;

    // ---- workspace carve-up (all offsets 256B aligned) ----
    char* ws = (char*)d_ws;
    size_t off = 0;
    auto alloc = [&](size_t bytes) -> char* {
        char* p = ws + off;
        off = align256(off + bytes);
        return p;
    };
    unsigned short* Wb_s1 = (unsigned short*)alloc((size_t)DH    * DIN   * 2);
    unsigned short* Wb_s2 = (unsigned short*)alloc((size_t)DH    * DH    * 2);
    unsigned short* Wb_s3 = (unsigned short*)alloc((size_t)DH    * DH    * 2);
    unsigned short* Wb_s4 = (unsigned short*)alloc((size_t)DH    * DH    * 2);
    unsigned short* Wb_m1 = (unsigned short*)alloc((size_t)DOUTC * DH    * 2);
    unsigned short* Wb_m2 = (unsigned short*)alloc((size_t)DOUTC * DOUTC * 2);
    unsigned short* Wb_l1 = (unsigned short*)alloc((size_t)DOUTC * DH    * 2);
    unsigned short* Wb_l2 = (unsigned short*)alloc((size_t)DOUTC * DOUTC * 2);
    float*          agg0  = (float*)         alloc((size_t)NNODES * DIN * 4);
    unsigned short* agg0b = (unsigned short*)alloc((size_t)NNODES * DIN * 2);
    unsigned short* ta    = (unsigned short*)alloc((size_t)NNODES * DH  * 2);
    unsigned short* tb    = (unsigned short*)alloc((size_t)NNODES * DH  * 2);
    float*          hf    = (float*)         alloc((size_t)NNODES * DH  * 4);
    float*          agg1  = (float*)         alloc((size_t)NNODES * DH  * 4);
    unsigned short* th    = (unsigned short*)alloc((size_t)NNODES * DOUTC * 2);

    const int TB = 256;
    auto cvt = [&](const float* src, unsigned short* dst, int n) {
        k_cvt_f32_bf16<<<(n + TB - 1) / TB, TB, 0, stream>>>(src, dst, n);
    };
    auto gemm = [&](const unsigned short* Abf, const unsigned short* Wbf,
                    const float* bias, float* oF, unsigned short* oB,
                    int M, int K, int Dn, int act) {
        dim3 grid((M + 127) / 128, Dn / 64);
        k_gemm_bias_act<<<grid, TB, 0, stream>>>(
            (const __bf16*)Abf, (const __bf16*)Wbf, bias, oF, oB, M, K, Dn, act);
    };

    // 1) weights -> bf16
    cvt(W_s1, Wb_s1, DH * DIN);
    cvt(W_s2, Wb_s2, DH * DH);
    cvt(W_s3, Wb_s3, DH * DH);
    cvt(W_s4, Wb_s4, DH * DH);
    cvt(W_m1, Wb_m1, DOUTC * DH);
    cvt(W_m2, Wb_m2, DOUTC * DOUTC);
    cvt(W_l1, Wb_l1, DOUTC * DH);
    cvt(W_l2, Wb_l2, DOUTC * DOUTC);

    // 2) agg0 = x + scatter_sum(x[src] -> dst)
    {
        int n4 = NNODES * DIN / 4;
        k_copy_f32x4<<<(n4 + TB - 1) / TB, TB, 0, stream>>>(x, agg0, n4);
        long long work = (long long)NEDGES * (DIN / 4);
        k_scatter_add4<<<(unsigned)((work + TB - 1) / TB), TB, 0, stream>>>(
            x, ei, agg0, NEDGES, DIN);
        cvt(agg0, agg0b, NNODES * DIN);
    }

    // 3) mlp_shared: 64->512 lrelu, 512->512 lrelu x2, 512->512 then relu (fused)
    gemm(agg0b, Wb_s1, b_s1, nullptr, ta, NNODES, DIN, DH, 1);
    gemm(ta,    Wb_s2, b_s2, nullptr, tb, NNODES, DH,  DH, 1);
    gemm(tb,    Wb_s3, b_s3, nullptr, ta, NNODES, DH,  DH, 1);
    gemm(ta,    Wb_s4, b_s4, hf, nullptr, NNODES, DH,  DH, 2);   // relu, f32 h

    // 4) agg1 = h + scatter_sum(h[src] -> dst)
    {
        int n4 = NNODES * DH / 4;
        k_copy_f32x4<<<(n4 + TB - 1) / TB, TB, 0, stream>>>(hf, agg1, n4);
        long long work = (long long)NEDGES * (DH / 4);
        k_scatter_add4<<<(unsigned)((work + TB - 1) / TB), TB, 0, stream>>>(
            hf, ei, agg1, NEDGES, DH);
        cvt(agg1, ta, NNODES * DH);   // reuse ta as bf16 agg1
    }

    // 5) mu head: relu(agg1 @ Wm1^T + b) @ Wm2^T + b  -> out[0 : N*128)
    gemm(ta, Wb_m1, b_m1, nullptr, th, NNODES, DH,    DOUTC, 2);
    gemm(th, Wb_m2, b_m2, out, nullptr, NNODES, DOUTC, DOUTC, 0);

    // 6) logvar head -> out[N*128 : 2*N*128)
    gemm(ta, Wb_l1, b_l1, nullptr, th, NNODES, DH,    DOUTC, 2);
    gemm(th, Wb_l2, b_l2, out + (size_t)NNODES * DOUTC, nullptr,
         NNODES, DOUTC, DOUTC, 0);
}
